// PointNet2SemSeg_23184233464113
// MI455X (gfx1250) — compile-verified, matrix operand order checked by reading, not run on hardware
//
#include <hip/hip_runtime.h>
#include <math.h>

typedef __attribute__((ext_vector_type(16))) _Float16 v16h;
typedef __attribute__((ext_vector_type(8)))  _Float16 v8h;
typedef __attribute__((ext_vector_type(8)))  float    v8f;
typedef long long ll;

static constexpr float kBnScale = 0.99999500003749971f; // 1/sqrt(1+1e-5)

// ---------------------------------------------------------------- utilities
__global__ void k_zero_f32(float* __restrict__ p, ll n) {
  ll stride = (ll)gridDim.x * blockDim.x;
  for (ll i = (ll)blockIdx.x * blockDim.x + threadIdx.x; i < n; i += stride) p[i] = 0.f;
}

// (B,C,N) -> (B,N,C)
__global__ void k_transpose(const float* __restrict__ f, int B, int C, int Np,
                            float* __restrict__ out) {
  ll total = (ll)B * Np * C;
  ll stride = (ll)gridDim.x * blockDim.x;
  for (ll i = (ll)blockIdx.x * blockDim.x + threadIdx.x; i < total; i += stride) {
    ll b = i / ((ll)Np * C);
    ll rem = i - b * (ll)Np * C;
    int n = (int)(rem / C), c = (int)(rem % C);
    out[i] = f[(b * C + c) * (ll)Np + n];
  }
}

// skip0 = concat(xyz, feats0) -> (rows, 3+C) f32
__global__ void k_skip0(const float* __restrict__ xyz, const float* __restrict__ feats0,
                        int C, ll rows, float* __restrict__ out) {
  int K = 3 + C;
  ll total = rows * K;
  ll stride = (ll)gridDim.x * blockDim.x;
  for (ll i = (ll)blockIdx.x * blockDim.x + threadIdx.x; i < total; i += stride) {
    ll row = i / K; int k = (int)(i % K);
    out[i] = (k < 3) ? xyz[row * 3 + k] : feats0[row * C + (k - 3)];
  }
}

// ------------------------------------------------------------ FPS (1 block/batch)
__global__ void k_fps(const float* __restrict__ xyz, int Np, int S,
                      int* __restrict__ idx, float* __restrict__ newxyz) {
  int b = blockIdx.x, tid = threadIdx.x;
  __shared__ float s_red[256];
  __shared__ int   s_idx[256];
  __shared__ float s_c[3];
  const float* X = xyz + (ll)b * Np * 3;
  float mind[32];
  int per = (Np + 255) >> 8;
  for (int t = 0; t < per; ++t) mind[t] = 1e10f;
  int far = 0;
  for (int i = 0; i < S; ++i) {
    if (tid == 0) {
      idx[b * S + i] = far;
      float cx = X[far * 3], cy = X[far * 3 + 1], cz = X[far * 3 + 2];
      s_c[0] = cx; s_c[1] = cy; s_c[2] = cz;
      float* nz = newxyz + ((ll)b * S + i) * 3;
      nz[0] = cx; nz[1] = cy; nz[2] = cz;
    }
    __syncthreads();
    float cx = s_c[0], cy = s_c[1], cz = s_c[2];
    float bm = -1.0f; int bi = Np;
    for (int t = 0; t < per; ++t) {
      int p = t * 256 + tid;
      if (p < Np) {
        float dx = X[p * 3] - cx, dy = X[p * 3 + 1] - cy, dz = X[p * 3 + 2] - cz;
        float d = dx * dx + dy * dy + dz * dz;
        float m = fminf(mind[t], d);
        mind[t] = m;
        if (m > bm || (m == bm && p < bi)) { bm = m; bi = p; }
      }
    }
    s_red[tid] = bm; s_idx[tid] = bi;
    __syncthreads();
    for (int s2 = 128; s2 > 0; s2 >>= 1) {
      if (tid < s2) {
        float o = s_red[tid + s2];
        if (o > s_red[tid] || (o == s_red[tid] && s_idx[tid + s2] < s_idx[tid])) {
          s_red[tid] = o; s_idx[tid] = s_idx[tid + s2];
        }
      }
      __syncthreads();
    }
    far = s_idx[0];
    __syncthreads();
  }
}

// --------------------------------------------------- ball query (1 wave / query)
__global__ void k_ball(const float* __restrict__ xyz, const float* __restrict__ newxyz,
                       int Np, int S, float r2, int ns, int* __restrict__ nidx) {
  int q = blockIdx.x;            // b*S + s
  int b = q / S;
  int lane = threadIdx.x;
  float cx = newxyz[q * 3], cy = newxyz[q * 3 + 1], cz = newxyz[q * 3 + 2];
  const float* X = xyz + (ll)b * Np * 3;
  int cnt = 0, first = -1;
  for (int base = 0; base < Np && cnt < ns; base += 32) {
    int p = base + lane;
    bool pred = false;
    if (p < Np) {
      float dx = X[p * 3] - cx, dy = X[p * 3 + 1] - cy, dz = X[p * 3 + 2] - cz;
      pred = (dx * dx + dy * dy + dz * dz) < r2;
    }
    unsigned mask = (unsigned)__ballot(pred);
    int off = __popc(mask & ((1u << lane) - 1u));
    if (pred && cnt + off < ns) nidx[(ll)q * ns + cnt + off] = p;
    if (first < 0 && mask) first = base + __builtin_ctz(mask);
    cnt += __popc(mask);
  }
  if (cnt > ns) cnt = ns;
  if (first < 0) first = 0;
  for (int i = cnt + lane; i < ns; i += 32) nidx[(ll)q * ns + i] = first;
}

// ------------------------------------- build grouped fp16 GEMM input (rows,Kpad)
__global__ void k_group_pack(const float* __restrict__ xyz, const float* __restrict__ feats,
                             const float* __restrict__ newxyz, const int* __restrict__ nidx,
                             int Np, int C, int S, int ns, int Kpad, ll total,
                             _Float16* __restrict__ out) {
  ll stride = (ll)gridDim.x * blockDim.x;
  for (ll i = (ll)blockIdx.x * blockDim.x + threadIdx.x; i < total; i += stride) {
    ll row = i / Kpad; int k = (int)(i % Kpad);
    ll q = row / ns;   int j = (int)(row % ns);
    int b = (int)(q / S);
    int p = nidx[q * ns + j];
    float v = 0.f;
    if (k < 3)            v = xyz[((ll)b * Np + p) * 3 + k] - newxyz[q * 3 + k];
    else if (k < 3 + C)   v = feats[((ll)b * Np + p) * C + (k - 3)];
    out[i] = (_Float16)v;
  }
}

// ---------------- weights: W(K,N) f32 -> Wt(Npad,Kpad) f16 + fused BN scale/shift
__global__ void k_prep_weights(const float* __restrict__ W, const float* __restrict__ bias,
                               const float* __restrict__ g, const float* __restrict__ beta,
                               int K, int N, int Kpad, int Npad,
                               _Float16* __restrict__ Wt, float* __restrict__ scale,
                               float* __restrict__ shift) {
  ll total = (ll)Kpad * Npad;
  ll stride = (ll)gridDim.x * blockDim.x;
  for (ll i = (ll)blockIdx.x * blockDim.x + threadIdx.x; i < total; i += stride) {
    int n = (int)(i / Kpad), k = (int)(i % Kpad);
    float w = (k < K && n < N) ? W[(ll)k * N + n] : 0.f;
    Wt[i] = (_Float16)w;
    if (k == 0) {
      if (n < N) { float s = g[n] * kBnScale; scale[n] = s; shift[n] = s * bias[n] + beta[n]; }
      else       { scale[n] = 0.f; shift[n] = 0.f; }
    }
  }
}

// ---------------- WMMA GEMM + BN + ReLU (+optional fused neighborhood max-pool)
// A: (M,Kpad) f16 row-major; Bt: (Npad,Kpad) f16 (W transposed).
// Register blocking: each wave computes 1 M-tile x 4 N-tiles so the streaming A
// fragment feeds 4 WMMAs (A is the HBM-bound operand; Bt is L2-resident).
// N-tile tails are clamped: duplicate sub-tiles recompute/store identical data.
__global__ void k_wmma_gemm(const _Float16* __restrict__ A, int lda,
                            const _Float16* __restrict__ Bt, int ldb,
                            const float* __restrict__ scale, const float* __restrict__ shift,
                            int Kpad, int Npad,
                            _Float16* __restrict__ out16, int ld16,
                            float* __restrict__ out32, int ld32, int Nexact,
                            float* __restrict__ pool, int pool_ns, int pool_stride,
                            int pool_coff) {
  int tm = blockIdx.x * 4 + threadIdx.y;   // grid.x = (M/16)/4, always exact here
  int tilesN = Npad >> 4;
  int tn0 = blockIdx.y * 4;
  int lane = threadIdx.x;
  int hh = lane >> 4;          // half-wave selector (ISA 16-bit operand layout)
  int r  = lane & 15;
  int koff = hh * 8;
  const _Float16* Ar = A + (ll)(tm * 16 + r) * lda;
  const _Float16* Br[4];
  int tns[4];
#pragma unroll
  for (int t = 0; t < 4; ++t) {
    int tn = tn0 + t;
    if (tn >= tilesN) tn = tilesN - 1;     // clamp: harmless duplicate work
    tns[t] = tn;
    Br[t] = Bt + (ll)(tn * 16 + r) * ldb;
  }
  v8f acc[4];
#pragma unroll
  for (int t = 0; t < 4; ++t) acc[t] = (v8f){0.f, 0.f, 0.f, 0.f, 0.f, 0.f, 0.f, 0.f};

  for (int k = 0; k < Kpad; k += 32) {
    union { v16h v; v8h h[2]; } a;
    a.h[0] = *reinterpret_cast<const v8h*>(Ar + k + koff);
    a.h[1] = *reinterpret_cast<const v8h*>(Ar + k + 16 + koff);
    __builtin_prefetch((const void*)(Ar + k + 512), 0, 1);  // global_prefetch_b8
#pragma unroll
    for (int t = 0; t < 4; ++t) {
      union { v16h v; v8h h[2]; } b;
      b.h[0] = *reinterpret_cast<const v8h*>(Br[t] + k + koff);
      b.h[1] = *reinterpret_cast<const v8h*>(Br[t] + k + 16 + koff);
      acc[t] = __builtin_amdgcn_wmma_f32_16x16x32_f16(false, a.v, false, b.v,
                                                      (short)0, acc[t], false, false);
    }
  }

#pragma unroll
  for (int t = 0; t < 4; ++t) {
    int col = tns[t] * 16 + r;
    float sc = scale[col], sh = shift[col];
#pragma unroll
    for (int i = 0; i < 8; ++i) {
      int row = tm * 16 + i + hh * 8;   // C/D layout: VGPR i -> M = i + 8*(lane>=16)
      float v = fmaxf(acc[t][i] * sc + sh, 0.0f);
      if (out16) out16[(ll)row * ld16 + col] = (_Float16)v;
      if (out32 && col < Nexact) out32[(ll)row * ld32 + col] = v;
      if (pool && col < Nexact) {
        // relu output >= 0 -> float ordering == u32 ordering; buffer pre-zeroed
        unsigned* p = reinterpret_cast<unsigned*>(pool) +
                      (ll)(row / pool_ns) * pool_stride + pool_coff + col;
        atomicMax(p, __float_as_uint(v));
      }
    }
  }
}

// ---------------------------------------------- f32 (rows,C) -> f16 (rows,Kpad)
__global__ void k_pack_f16(const float* __restrict__ in, int C, int Kpad, ll total,
                           _Float16* __restrict__ out) {
  ll stride = (ll)gridDim.x * blockDim.x;
  for (ll i = (ll)blockIdx.x * blockDim.x + threadIdx.x; i < total; i += stride) {
    ll row = i / Kpad; int k = (int)(i % Kpad);
    out[i] = (_Float16)((k < C) ? in[row * C + k] : 0.f);
  }
}

// ------------------------------------------------------ 3-NN (1 thread / query)
__global__ void k_knn3(const float* __restrict__ unk, const float* __restrict__ kn,
                       int Su, int Sk, int B, int* __restrict__ idx3,
                       float* __restrict__ w3) {
  int q = blockIdx.x * blockDim.x + threadIdx.x;
  if (q >= B * Su) return;
  int b = q / Su;
  float ux = unk[q * 3], uy = unk[q * 3 + 1], uz = unk[q * 3 + 2];
  const float* K = kn + (ll)b * Sk * 3;
  float d0 = 1e30f, d1 = 1e30f, d2 = 1e30f; int i0 = 0, i1 = 0, i2 = 0;
  for (int p = 0; p < Sk; ++p) {
    float dx = K[p * 3] - ux, dy = K[p * 3 + 1] - uy, dz = K[p * 3 + 2] - uz;
    float d = dx * dx + dy * dy + dz * dz;
    if (d < d0)      { d2 = d1; i2 = i1; d1 = d0; i1 = i0; d0 = d; i0 = p; }
    else if (d < d1) { d2 = d1; i2 = i1; d1 = d; i1 = p; }
    else if (d < d2) { d2 = d; i2 = p; }
  }
  float w0 = 1.f / (sqrtf(fmaxf(d0, 0.f)) + 1e-8f);
  float w1 = 1.f / (sqrtf(fmaxf(d1, 0.f)) + 1e-8f);
  float w2 = 1.f / (sqrtf(fmaxf(d2, 0.f)) + 1e-8f);
  float ws = w0 + w1 + w2;
  idx3[q * 3] = i0; idx3[q * 3 + 1] = i1; idx3[q * 3 + 2] = i2;
  w3[q * 3] = w0 / ws; w3[q * 3 + 1] = w1 / ws; w3[q * 3 + 2] = w2 / ws;
}

// --------------------------- interp(known_feats) ++ skip -> fp16 GEMM input
__global__ void k_interp_pack(const int* __restrict__ idx3, const float* __restrict__ w3,
                              const float* __restrict__ kf, int Ck, int Sk,
                              const float* __restrict__ skip, int Cs,
                              int Su, int Kpad, ll total, _Float16* __restrict__ out) {
  ll stride = (ll)gridDim.x * blockDim.x;
  for (ll i = (ll)blockIdx.x * blockDim.x + threadIdx.x; i < total; i += stride) {
    ll row = i / Kpad; int k = (int)(i % Kpad);
    float v = 0.f;
    if (k < Ck) {
      int b = (int)(row / Su);
      const float* kfb = kf + (ll)b * Sk * Ck;
      v = w3[row * 3 + 0] * kfb[(ll)idx3[row * 3 + 0] * Ck + k]
        + w3[row * 3 + 1] * kfb[(ll)idx3[row * 3 + 1] * Ck + k]
        + w3[row * 3 + 2] * kfb[(ll)idx3[row * 3 + 2] * Ck + k];
    } else if (k < Ck + Cs) {
      v = skip[row * Cs + (k - Ck)];
    }
    out[i] = (_Float16)v;
  }
}

// ------------------- attention head: a=sigmoid(row.W1+b1); l3 *= a  (1 wave/row)
__global__ void k_att_head(const _Float16* __restrict__ a16, const float* __restrict__ W1,
                           const float* __restrict__ b1, const float* __restrict__ l3,
                           float* __restrict__ out, int Ka, int C) {
  int row = blockIdx.x, lane = threadIdx.x;
  float p = 0.f;
  for (int k = lane; k < Ka; k += 32) p += (float)a16[(ll)row * Ka + k] * W1[k];
  for (int off = 16; off > 0; off >>= 1) p += __shfl_down(p, off, 32);
  float gsig = __shfl(p, 0, 32);
  gsig = 1.f / (1.f + expf(-(gsig + b1[0])));
  for (int c = lane; c < C; c += 32)
    out[(ll)row * C + c] = l3[(ll)row * C + c] * gsig;
}

// ------------------------------------------------- classifier final (K=128,N=2)
__global__ void k_cls_final(const _Float16* __restrict__ x, const float* __restrict__ W,
                            const float* __restrict__ b, float* __restrict__ out,
                            int rows, int K, int Ncls) {
  int row = blockIdx.x * blockDim.x + threadIdx.x;
  if (row >= rows) return;
  for (int n = 0; n < Ncls; ++n) {
    float acc = b[n];
    for (int k = 0; k < K; ++k) acc += (float)x[(ll)row * K + k] * W[(ll)k * Ncls + n];
    out[(ll)row * Ncls + n] = acc;
  }
}

// ============================================================================
extern "C" void kernel_launch(void* const* d_in, const int* in_sizes, int n_in,
                              void* d_out, int out_size, void* d_ws, size_t ws_size,
                              hipStream_t stream) {
  (void)in_sizes; (void)n_in; (void)out_size; (void)ws_size;
  const int B = 2, N0 = 8192, CIN = 8;
  const float* xyz      = (const float*)d_in[0];
  const float* features = (const float*)d_in[1];

  // ---- param layout: dict insertion order, each layer {W,b,g,beta} (4 arrays)
  struct LayerP { const float *W, *b, *g, *beta; int K, N; };
  auto LP = [&](int base, int K, int N) {
    return LayerP{(const float*)d_in[base], (const float*)d_in[base + 1],
                  (const float*)d_in[base + 2], (const float*)d_in[base + 3], K, N};
  };
  // sa1: 2,6,10 | 14,18,22 ; sa2: 26,30,34 | 38,42,46 ; sa3: 50,54,58
  // att: 62,66 ; fp3: 70,74 ; fp2: 78,82 ; fp1: 86,90 ; cls: 94,98

  auto pad32 = [](int x) { return (x + 31) & ~31; };
  auto nb = [](ll total, int t) { return (unsigned)((total + t - 1) / t); };

  // ---- workspace bump allocator (256B aligned)
  char* wsb = (char*)d_ws;
  size_t off = 0;
  auto alloc = [&](size_t bytes) {
    void* p = wsb + off;
    off = (off + bytes + 255) & ~(size_t)255;
    return p;
  };
  float*    feats0 = (float*)alloc((size_t)B * N0 * CIN * 4);
  float*    skip0  = (float*)alloc((size_t)B * N0 * 11 * 4);
  float*    l1xyz  = (float*)alloc((size_t)B * 2048 * 3 * 4);
  float*    l1     = (float*)alloc((size_t)B * 2048 * 256 * 4);
  float*    l2xyz  = (float*)alloc((size_t)B * 512 * 3 * 4);
  float*    l2     = (float*)alloc((size_t)B * 512 * 512 * 4);
  float*    l3xyz  = (float*)alloc((size_t)B * 128 * 3 * 4);
  float*    l3     = (float*)alloc((size_t)B * 128 * 1024 * 4);
  float*    l3att  = (float*)alloc((size_t)B * 128 * 1024 * 4);
  float*    l2new  = (float*)alloc((size_t)B * 512 * 512 * 4);
  float*    l1new  = (float*)alloc((size_t)B * 2048 * 256 * 4);
  _Float16* l0h    = (_Float16*)alloc((size_t)B * N0 * 128 * 2);
  _Float16* a16buf = (_Float16*)alloc((size_t)B * 128 * 512 * 2);
  int*      fpsidx = (int*)alloc((size_t)B * 2048 * 4);
  int*      nidx   = (int*)alloc((size_t)B * 2048 * 64 * 4);      // max B*S*ns
  int*      idx3   = (int*)alloc((size_t)B * N0 * 3 * 4);
  float*    w3     = (float*)alloc((size_t)B * N0 * 3 * 4);
  _Float16* wt     = (_Float16*)alloc((size_t)1536 * 512 * 2);    // max Kpad*Npad
  float*    scaleb = (float*)alloc(1024 * 4);
  float*    shiftb = (float*)alloc(1024 * 4);
  _Float16* A16    = (_Float16*)alloc((size_t)131072 * 288 * 2);  // 75.5 MB
  _Float16* ping   = (_Float16*)alloc((size_t)262144 * 64 * 2);   // 33.6 MB
  _Float16* pong   = (_Float16*)alloc((size_t)65536 * 512 * 2);   // 67.1 MB

  // ---- fused GEMM+BN+ReLU launcher (M/16 always divisible by 4 in this net)
  auto gemm = [&](const _Float16* Ain, int rows, int Kpad, LayerP L,
                  _Float16* o16, float* o32, float* pool, int pool_ns,
                  int pool_stride, int pool_coff) {
    int Npad = pad32(L.N);
    ll wtot = (ll)Kpad * Npad;
    k_prep_weights<<<nb(wtot, 256), 256, 0, stream>>>(L.W, L.b, L.g, L.beta, L.K, L.N,
                                                      Kpad, Npad, wt, scaleb, shiftb);
    dim3 grid((unsigned)(rows / 64), nb(Npad / 16, 4));
    dim3 blk(32, 4);
    k_wmma_gemm<<<grid, blk, 0, stream>>>(Ain, Kpad, wt, Kpad, scaleb, shiftb, Kpad, Npad,
                                          o16, o16 ? pad32(L.N) : 0,
                                          o32, o32 ? L.N : 0, L.N,
                                          pool, pool_ns, pool_stride, pool_coff);
  };

  // ---- set-abstraction stage
  auto run_sa = [&](const float* xyz_in, const float* feats_in, int Np, int C, int S,
                    int nbranch, const float* radii, const int* nss, const LayerP* Ls,
                    float* newxyz, float* fout, int Ctot) {
    k_fps<<<B, 256, 0, stream>>>(xyz_in, Np, S, fpsidx, newxyz);
    ll ztot = (ll)B * S * Ctot;
    k_zero_f32<<<nb(ztot, 256), 256, 0, stream>>>(fout, ztot);
    int coff = 0;
    for (int br = 0; br < nbranch; ++br) {
      int ns = nss[br];
      float r = radii[br];
      k_ball<<<B * S, 32, 0, stream>>>(xyz_in, newxyz, Np, S, r * r, ns, nidx);
      int Kp0 = pad32(3 + C);
      int rows = B * S * ns;
      ll gtot = (ll)rows * Kp0;
      k_group_pack<<<nb(gtot, 256), 256, 0, stream>>>(xyz_in, feats_in, newxyz, nidx,
                                                      Np, C, S, ns, Kp0, gtot, A16);
      LayerP L0 = Ls[br * 3 + 0], L1v = Ls[br * 3 + 1], L2v = Ls[br * 3 + 2];
      gemm(A16,  rows, Kp0,           L0, ping, nullptr, nullptr, 0, 0, 0);
      gemm(ping, rows, pad32(L0.N),   L1v, pong, nullptr, nullptr, 0, 0, 0);
      gemm(pong, rows, pad32(L1v.N),  L2v, nullptr, nullptr, fout, ns, Ctot, coff);
      coff += L2v.N;
    }
  };

  // ---- feature-propagation stage
  auto run_fp = [&](const float* unk_xyz, int Su, const float* kn_xyz, int Sk,
                    const float* kf, int Ck, const float* skip, int Cs,
                    LayerP L0, LayerP L1v, _Float16* out16_final, float* out32_final) {
    int nq = B * Su;
    k_knn3<<<nb(nq, 128), 128, 0, stream>>>(unk_xyz, kn_xyz, Su, Sk, B, idx3, w3);
    int Kp = pad32(Ck + Cs);
    ll t = (ll)nq * Kp;
    k_interp_pack<<<nb(t, 256), 256, 0, stream>>>(idx3, w3, kf, Ck, Sk, skip, Cs,
                                                  Su, Kp, t, A16);
    gemm(A16,  nq, Kp,          L0,  ping, nullptr, nullptr, 0, 0, 0);
    gemm(ping, nq, pad32(L0.N), L1v, out16_final, out32_final, nullptr, 0, 0, 0);
  };

  // ================================ pipeline ================================
  ll t0 = (ll)B * N0 * CIN;
  k_transpose<<<nb(t0, 256), 256, 0, stream>>>(features, B, CIN, N0, feats0);
  k_skip0<<<nb((ll)B * N0 * 11, 256), 256, 0, stream>>>(xyz, feats0, CIN, (ll)B * N0, skip0);

  // SA1: N=8192 -> S=2048, branches (r=0.1,ns=32),(r=0.2,ns=64), Ctot=256
  {
    LayerP Ls[6] = {LP(2, 11, 64),  LP(6, 64, 64),  LP(10, 64, 128),
                    LP(14, 11, 64), LP(18, 64, 96), LP(22, 96, 128)};
    float rr[2] = {0.1f, 0.2f}; int nn[2] = {32, 64};
    run_sa(xyz, feats0, N0, CIN, 2048, 2, rr, nn, Ls, l1xyz, l1, 256);
  }
  // SA2: 2048 -> 512, (0.2,64),(0.4,128), Ctot=512
  {
    LayerP Ls[6] = {LP(26, 259, 128), LP(30, 128, 128), LP(34, 128, 256),
                    LP(38, 259, 128), LP(42, 128, 196), LP(46, 196, 256)};
    float rr[2] = {0.2f, 0.4f}; int nn[2] = {64, 128};
    run_sa(l1xyz, l1, 2048, 256, 512, 2, rr, nn, Ls, l2xyz, l2, 512);
  }
  // SA3: 512 -> 128, (0.8,256), Ctot=1024
  {
    LayerP Ls[3] = {LP(50, 515, 256), LP(54, 256, 512), LP(58, 512, 1024)};
    float rr[1] = {0.8f}; int nn[1] = {256};
    run_sa(l2xyz, l2, 512, 512, 128, 1, rr, nn, Ls, l3xyz, l3, 1024);
  }
  // attention: a = relu-BN(l3 @ att0); g = sigmoid(a @ att1); l3att = l3 * g
  {
    ll t = (ll)B * 128 * 1024;
    k_pack_f16<<<nb(t, 256), 256, 0, stream>>>(l3, 1024, 1024, t, A16);
    gemm(A16, B * 128, 1024, LP(62, 1024, 512), a16buf, nullptr, nullptr, 0, 0, 0);
    k_att_head<<<B * 128, 32, 0, stream>>>(a16buf, (const float*)d_in[66],
                                           (const float*)d_in[67], l3, l3att, 512, 1024);
  }
  // FP3 / FP2 / FP1
  run_fp(l2xyz, 512, l3xyz, 128, l3att, 1024, l2, 512,
         LP(70, 1536, 512), LP(74, 512, 512), nullptr, l2new);
  run_fp(l1xyz, 2048, l2xyz, 512, l2new, 512, l1, 256,
         LP(78, 768, 256), LP(82, 256, 256), nullptr, l1new);
  run_fp(xyz, N0, l1xyz, 2048, l1new, 256, skip0, 11,
         LP(86, 267, 128), LP(90, 128, 128), l0h, nullptr);

  // classifier: conv_bn_relu(128->128) then linear(128->2) -> d_out
  gemm(l0h, B * N0, 128, LP(94, 128, 128), ping, nullptr, nullptr, 0, 0, 0);
  k_cls_final<<<nb(B * N0, 128), 128, 0, stream>>>(ping, (const float*)d_in[98],
                                                   (const float*)d_in[99],
                                                   (float*)d_out, B * N0, 128, 2);
}
